// Spectra_60327110639874
// MI455X (gfx1250) — matrix operand order repaired and checked by reading, not compiled, not run on hardware
//
#include <hip/hip_runtime.h>
#include <hip/hip_bf16.h>

typedef __attribute__((ext_vector_type(16))) __bf16 v16bf;
typedef __attribute__((ext_vector_type(8)))  float  v8f;

#define PATCH 16
#define HEADS 4
#define DHEAD 64
#define D_MODEL 512
#define PD 1024
#define HID 256
#define TLEN 1025
#define BATCH 8
#define RTOT (BATCH * TLEN)   // 8200 rows

__device__ __forceinline__ float sigmoidf_(float x) { return 1.f / (1.f + __expf(-x)); }

__device__ __forceinline__ unsigned int pack_bf16x2(float a, float b)
{
    unsigned short lo = __builtin_bit_cast(unsigned short, (__bf16)a);
    unsigned short hi = __builtin_bit_cast(unsigned short, (__bf16)b);
    return (unsigned int)lo | ((unsigned int)hi << 16);
}

__device__ __forceinline__ v16bf load_a_frag(const float* p0)
{
    float4 a0 = *(const float4*)(p0);
    float4 a1 = *(const float4*)(p0 + 4);
    float4 a2 = *(const float4*)(p0 + 16);
    float4 a3 = *(const float4*)(p0 + 20);
    v16bf f;
    f[0]  = (__bf16)a0.x; f[1]  = (__bf16)a0.y; f[2]  = (__bf16)a0.z; f[3]  = (__bf16)a0.w;
    f[4]  = (__bf16)a1.x; f[5]  = (__bf16)a1.y; f[6]  = (__bf16)a1.z; f[7]  = (__bf16)a1.w;
    f[8]  = (__bf16)a2.x; f[9]  = (__bf16)a2.y; f[10] = (__bf16)a2.z; f[11] = (__bf16)a2.w;
    f[12] = (__bf16)a3.x; f[13] = (__bf16)a3.y; f[14] = (__bf16)a3.z; f[15] = (__bf16)a3.w;
    return f;
}

__device__ __forceinline__ v16bf load_a_frag_mul(const float* p0, const float* q0)
{
    float4 a0 = *(const float4*)(p0);
    float4 a1 = *(const float4*)(p0 + 4);
    float4 a2 = *(const float4*)(p0 + 16);
    float4 a3 = *(const float4*)(p0 + 20);
    float4 u0 = *(const float4*)(q0);
    float4 u1 = *(const float4*)(q0 + 4);
    float4 u2 = *(const float4*)(q0 + 16);
    float4 u3 = *(const float4*)(q0 + 20);
    v16bf f;
    f[0]  = (__bf16)(a0.x * u0.x); f[1]  = (__bf16)(a0.y * u0.y);
    f[2]  = (__bf16)(a0.z * u0.z); f[3]  = (__bf16)(a0.w * u0.w);
    f[4]  = (__bf16)(a1.x * u1.x); f[5]  = (__bf16)(a1.y * u1.y);
    f[6]  = (__bf16)(a1.z * u1.z); f[7]  = (__bf16)(a1.w * u1.w);
    f[8]  = (__bf16)(a2.x * u2.x); f[9]  = (__bf16)(a2.y * u2.y);
    f[10] = (__bf16)(a2.z * u2.z); f[11] = (__bf16)(a2.w * u2.w);
    f[12] = (__bf16)(a3.x * u3.x); f[13] = (__bf16)(a3.y * u3.y);
    f[14] = (__bf16)(a3.z * u3.z); f[15] = (__bf16)(a3.w * u3.w);
    return f;
}

// ---------------------------------------------------------------------------
// WMMA GEMM: out[M,N] = act( scale*(A (*Amul) @ W) + bias ) + resid
// Block = 256 threads (8 waves) -> 256(M) x 64(N); each wave owns 32x64.
// W tiles (32x64, bf16, K-transposed) double-buffered in LDS. Staging: each
// thread owns one column, loads K-adjacent pairs, packs bf16x2, stores 4
// ds_store_b32 at constant offsets. act: 0=none, 1=silu, 2=sigmoid
// ---------------------------------------------------------------------------
template <bool HAS_MUL>
__global__ void gemm_wmma(const float* __restrict__ A, const float* __restrict__ Amul,
                          const float* __restrict__ W, const float* __restrict__ bias,
                          const float* __restrict__ resid, float* __restrict__ out,
                          int M, int N, int K, float scale, int act)
{
    __shared__ __bf16 Ws[2][64 * 32];       // [buf][col*32 + p], p = k with k 8..15 and 16..23 swapped; 2x4KB
    int tid  = threadIdx.x;                 // 256
    int wid  = tid >> 5, lane = tid & 31;
    int half = lane >> 4, idx = lane & 15;

    int n0 = blockIdx.y << 6;               // 64-col strip
    int m0 = (blockIdx.x << 8) + (wid << 5);// wave covers 32 rows

    int ar0 = m0 + idx;
    int ar1 = ar0 + 16;
    int ar0c = ar0 < M ? ar0 : (M - 1);
    int ar1c = ar1 < M ? ar1 : (M - 1);
    const float* Arow0 = A + (size_t)ar0c * K;
    const float* Arow1 = A + (size_t)ar1c * K;
    const float* Mrow0 = HAS_MUL ? (Amul + (size_t)ar0c * K) : nullptr;
    const float* Mrow1 = HAS_MUL ? (Amul + (size_t)ar1c * K) : nullptr;

    // staging geometry: col sc fixed per thread; 4 k-pairs per thread
    int sc  = tid & 63;                     // column 0..63
    int rpb = tid >> 6;                     // k-pair base 0..3 (pairs rpb+4*i)
    const float* Wcol = W + (n0 + sc);
    unsigned int* ws32 = (unsigned int*)Ws; // buffer b at offset b*1024 dwords
    int sbase = sc * 16 + rpb;              // uint32 index; +0,8,4,12 for i=0..3

    // ---- prologue: stage tile k0=0 into buffer 0 ----
#pragma unroll
    for (int i = 0; i < 4; ++i) {
        int rp = rpb + 4 * i;
        float w0 = Wcol[(size_t)(2 * rp) * N];
        float w1 = Wcol[(size_t)(2 * rp + 1) * N];
        ws32[sbase + 4 * (((i & 1) << 1) | (i >> 1))] = pack_bf16x2(w0, w1);
    }
    __syncthreads();

    const int nsteps = K >> 5;
    v8f acc[8] = {};                        // [mt*4 + nt]

    for (int it = 0; it < nsteps; ++it) {
        int k0 = it << 5;
        int cur = it & 1, nxt = cur ^ 1;

        // 1. issue next W tile's global loads (held in 8 regs)
        float w0r[4], w1r[4];
        if (it + 1 < nsteps) {
#pragma unroll
            for (int i = 0; i < 4; ++i) {
                int rp = rpb + 4 * i;
                w0r[i] = Wcol[(size_t)(k0 + 32 + 2 * rp) * N];
                w1r[i] = Wcol[(size_t)(k0 + 32 + 2 * rp + 1) * N];
            }
        }

        // 2. A fragments
        int ks0 = k0 + (half << 3);
        v16bf af0 = HAS_MUL ? load_a_frag_mul(Arow0 + ks0, Mrow0 + ks0)
                            : load_a_frag(Arow0 + ks0);
        v16bf af1 = HAS_MUL ? load_a_frag_mul(Arow1 + ks0, Mrow1 + ks0)
                            : load_a_frag(Arow1 + ks0);

        // 3. all 4 B fragments, then 8 WMMAs back-to-back
        v16bf bfr[4];
#pragma unroll
        for (int nt = 0; nt < 4; ++nt)
            bfr[nt] = *(const v16bf*)&Ws[cur][((nt << 4) + idx) * 32 + (half << 4)];
#pragma unroll
        for (int nt = 0; nt < 4; ++nt)
            acc[nt] = __builtin_amdgcn_wmma_f32_16x16x32_bf16(false, af0, false, bfr[nt], (short)0, acc[nt], false, false);
#pragma unroll
        for (int nt = 0; nt < 4; ++nt)
            acc[4 + nt] = __builtin_amdgcn_wmma_f32_16x16x32_bf16(false, af1, false, bfr[nt], (short)0, acc[4 + nt], false, false);

        // 4. pack + commit next tile, then barrier
        if (it + 1 < nsteps) {
#pragma unroll
            for (int i = 0; i < 4; ++i)
                ws32[(nxt << 10) + sbase + 4 * (((i & 1) << 1) | (i >> 1))] = pack_bf16x2(w0r[i], w1r[i]);
        }
        __syncthreads();
    }

    // ---- epilogue ----
#pragma unroll
    for (int mt = 0; mt < 2; ++mt) {
#pragma unroll
        for (int nt = 0; nt < 4; ++nt) {
#pragma unroll
            for (int v = 0; v < 8; ++v) {
                int row = m0 + (mt << 4) + v + (half << 3);
                if (row >= M) continue;
                int col = n0 + (nt << 4) + idx;
                float xv = acc[mt * 4 + nt][v] * scale;
                if (bias) xv += bias[col];
                if (act == 1)      xv = xv * sigmoidf_(xv);
                else if (act == 2) xv = sigmoidf_(xv);
                size_t o = (size_t)row * N + col;
                if (resid) xv += resid[o];
                out[o] = xv;
            }
        }
    }
}

// ---------------------------------------------------------------------------
// Patch-embed GEMM: tok[b, p+1, :] = patches[b,p,:] @ W_e + b_e, masked -> 0
// patches[b,p,d] = x[b, (p%512)*32 + d/16, (p/512)*16 + d%16]
// M=8192, K=N=512; same 8-wave double-buffered structure.
// ---------------------------------------------------------------------------
__global__ void gemm_embed(const float* __restrict__ x, const float* __restrict__ We,
                           const float* __restrict__ be, const int* __restrict__ mask,
                           float* __restrict__ tok)
{
    const int N = D_MODEL, K = D_MODEL;
    __shared__ __bf16 Ws[2][64 * 32];
    int tid  = threadIdx.x;                 // 256
    int wid  = tid >> 5, lane = tid & 31;
    int half = lane >> 4, idx = lane & 15;

    int n0 = blockIdx.y << 6;
    int m0 = (blockIdx.x << 8) + (wid << 5);

    int ar0 = m0 + idx, ar1 = ar0 + 16;
    int b0_ = ar0 >> 10, p0_ = ar0 & 1023;
    int b1_ = ar1 >> 10, p1_ = ar1 & 1023;
    const float* xb0 = x + ((size_t)b0_ * 16384) * 32 + (p0_ >> 9) * 16 + (size_t)(p0_ & 511) * 32 * 32;
    const float* xb1 = x + ((size_t)b1_ * 16384) * 32 + (p1_ >> 9) * 16 + (size_t)(p1_ & 511) * 32 * 32;

    int sc  = tid & 63;
    int rpb = tid >> 6;
    const float* Wcol = We + (n0 + sc);
    unsigned int* ws32 = (unsigned int*)Ws;
    int sbase = sc * 16 + rpb;

#pragma unroll
    for (int i = 0; i < 4; ++i) {
        int rp = rpb + 4 * i;
        float w0 = Wcol[(size_t)(2 * rp) * N];
        float w1 = Wcol[(size_t)(2 * rp + 1) * N];
        ws32[sbase + 4 * (((i & 1) << 1) | (i >> 1))] = pack_bf16x2(w0, w1);
    }
    __syncthreads();

    const int nsteps = K >> 5;
    v8f acc[8] = {};

    for (int it = 0; it < nsteps; ++it) {
        int k0 = it << 5;
        int cur = it & 1, nxt = cur ^ 1;

        float w0r[4], w1r[4];
        if (it + 1 < nsteps) {
#pragma unroll
            for (int i = 0; i < 4; ++i) {
                int rp = rpb + 4 * i;
                w0r[i] = Wcol[(size_t)(k0 + 32 + 2 * rp) * N];
                w1r[i] = Wcol[(size_t)(k0 + 32 + 2 * rp + 1) * N];
            }
        }

        // A: element k -> xb[(k/16)*32 + k%16]; two in-group 8-float segments
        int ks0 = k0 + (half << 3);
        int ks1 = ks0 + 16;
        v16bf af0, af1;
        {
            const float* s0p = xb0 + (ks0 >> 4) * 32 + (ks0 & 15);
            const float* s1p = xb0 + (ks1 >> 4) * 32 + (ks1 & 15);
            float4 a0 = *(const float4*)(s0p);
            float4 a1 = *(const float4*)(s0p + 4);
            float4 a2 = *(const float4*)(s1p);
            float4 a3 = *(const float4*)(s1p + 4);
            af0[0]  = (__bf16)a0.x; af0[1]  = (__bf16)a0.y; af0[2]  = (__bf16)a0.z; af0[3]  = (__bf16)a0.w;
            af0[4]  = (__bf16)a1.x; af0[5]  = (__bf16)a1.y; af0[6]  = (__bf16)a1.z; af0[7]  = (__bf16)a1.w;
            af0[8]  = (__bf16)a2.x; af0[9]  = (__bf16)a2.y; af0[10] = (__bf16)a2.z; af0[11] = (__bf16)a2.w;
            af0[12] = (__bf16)a3.x; af0[13] = (__bf16)a3.y; af0[14] = (__bf16)a3.z; af0[15] = (__bf16)a3.w;
        }
        {
            const float* s0p = xb1 + (ks0 >> 4) * 32 + (ks0 & 15);
            const float* s1p = xb1 + (ks1 >> 4) * 32 + (ks1 & 15);
            float4 a0 = *(const float4*)(s0p);
            float4 a1 = *(const float4*)(s0p + 4);
            float4 a2 = *(const float4*)(s1p);
            float4 a3 = *(const float4*)(s1p + 4);
            af1[0]  = (__bf16)a0.x; af1[1]  = (__bf16)a0.y; af1[2]  = (__bf16)a0.z; af1[3]  = (__bf16)a0.w;
            af1[4]  = (__bf16)a1.x; af1[5]  = (__bf16)a1.y; af1[6]  = (__bf16)a1.z; af1[7]  = (__bf16)a1.w;
            af1[8]  = (__bf16)a2.x; af1[9]  = (__bf16)a2.y; af1[10] = (__bf16)a2.z; af1[11] = (__bf16)a2.w;
            af1[12] = (__bf16)a3.x; af1[13] = (__bf16)a3.y; af1[14] = (__bf16)a3.z; af1[15] = (__bf16)a3.w;
        }

        v16bf bfr[4];
#pragma unroll
        for (int nt = 0; nt < 4; ++nt)
            bfr[nt] = *(const v16bf*)&Ws[cur][((nt << 4) + idx) * 32 + (half << 4)];
#pragma unroll
        for (int nt = 0; nt < 4; ++nt)
            acc[nt] = __builtin_amdgcn_wmma_f32_16x16x32_bf16(false, af0, false, bfr[nt], (short)0, acc[nt], false, false);
#pragma unroll
        for (int nt = 0; nt < 4; ++nt)
            acc[4 + nt] = __builtin_amdgcn_wmma_f32_16x16x32_bf16(false, af1, false, bfr[nt], (short)0, acc[4 + nt], false, false);

        if (it + 1 < nsteps) {
#pragma unroll
            for (int i = 0; i < 4; ++i)
                ws32[(nxt << 10) + sbase + 4 * (((i & 1) << 1) | (i >> 1))] = pack_bf16x2(w0r[i], w1r[i]);
        }
        __syncthreads();
    }

    int mk0 = mask[0], mk1 = mask[1], mk2 = mask[2], mk3 = mask[3];
#pragma unroll
    for (int mt = 0; mt < 2; ++mt) {
#pragma unroll
        for (int nt = 0; nt < 4; ++nt) {
#pragma unroll
            for (int v = 0; v < 8; ++v) {
                int row = m0 + (mt << 4) + v + (half << 3);
                int rb = row >> 10, rp = row & 1023;
                int col = n0 + (nt << 4) + idx;
                float val = acc[mt * 4 + nt][v] + be[col];
                if (rp == mk0 || rp == mk1 || rp == mk2 || rp == mk3) val = 0.f;
                tok[((size_t)rb * TLEN + rp + 1) * D_MODEL + col] = val;
            }
        }
    }
}

// ---------------------------------------------------------------------------
__global__ void fill_cls(float* __restrict__ tok)
{
    int t = blockIdx.x * blockDim.x + threadIdx.x;
    if (t < BATCH * D_MODEL) {
        int b = t >> 9, c = t & 511;
        tok[(size_t)b * TLEN * D_MODEL + c] = 0.f;
    }
}

// ---------------------------------------------------------------------------
// LayerNorm over last dim (512), one wave per row.
// ---------------------------------------------------------------------------
__global__ void ln_rows(const float* __restrict__ in, const float* __restrict__ g,
                        const float* __restrict__ bt, float* __restrict__ out, int R)
{
    int row = blockIdx.x * (blockDim.x >> 5) + (threadIdx.x >> 5);
    if (row >= R) return;
    int lane = threadIdx.x & 31;
    const float* p = in + (size_t)row * D_MODEL;
    float vals[16], s = 0.f, ss = 0.f;
#pragma unroll
    for (int e = 0; e < 16; ++e) {
        float v = p[lane + 32 * e];
        vals[e] = v; s += v; ss += v * v;
    }
#pragma unroll
    for (int m = 16; m; m >>= 1) { s += __shfl_xor(s, m, 32); ss += __shfl_xor(ss, m, 32); }
    float mu  = s * (1.f / 512.f);
    float var = ss * (1.f / 512.f) - mu * mu;
    float rs  = rsqrtf(var + 1e-5f);
    float* q = out + (size_t)row * D_MODEL;
#pragma unroll
    for (int e = 0; e < 16; ++e) {
        int c = lane + 32 * e;
        q[c] = (vals[e] - mu) * rs * g[c] + bt[c];
    }
}

// ---------------------------------------------------------------------------
// i/f gate projections (N=4 each): one thread per output.
// ---------------------------------------------------------------------------
__global__ void ifgate(const float* __restrict__ xu, const float* __restrict__ Wi,
                       const float* __restrict__ Wf, float* __restrict__ ipre,
                       float* __restrict__ fpre, int R)
{
    int t = blockIdx.x * blockDim.x + threadIdx.x;
    if (t >= R * 8) return;
    int row = t >> 3, j = t & 7, col = j & 3;
    const float* W = (j < 4) ? Wi : Wf;
    const float* xr = xu + (size_t)row * PD;
    float s = 0.f;
    for (int k = 0; k < PD; ++k) s += xr[k] * W[k * 4 + col];
    if (j < 4) ipre[row * 4 + col] = s; else fpre[row * 4 + col] = s;
}

// ---------------------------------------------------------------------------
// Sequential mLSTM scan: one block per (b, head); C[64][64] in registers.
// ---------------------------------------------------------------------------
__global__ void mlstm_scan(const float* __restrict__ q, const float* __restrict__ k,
                           const float* __restrict__ v, const float* __restrict__ ipre,
                           const float* __restrict__ fpre, float* __restrict__ h)
{
    int bh = blockIdx.x;
    int b = bh >> 2, hd = bh & 3;
    int tid = threadIdx.x;                 // 256 threads
    __shared__ float qs[64], ks[64], vs[64], nqs[64], partial[256];
    __shared__ float s_ig, s_fg, s_den;
    float C[16];
#pragma unroll
    for (int j = 0; j < 16; ++j) C[j] = 0.f;
    float nreg = 0.f;      // valid for tid < 64
    float mreg = 0.f;      // valid for tid == 192 (stabilizer init = 0)
    int rr = tid >> 2, cbase = (tid & 3) << 4;

    for (int t = 0; t < TLEN; ++t) {
        size_t base = ((size_t)(b * TLEN + t) * HEADS + hd) * DHEAD;
        if (tid < 64)        qs[tid]        = q[base + tid];
        else if (tid < 128)  ks[tid - 64]   = k[base + tid - 64];
        else if (tid < 192)  vs[tid - 128]  = v[base + tid - 128];
        else if (tid == 192) {
            size_t gb = (size_t)(b * TLEN + t) * HEADS + hd;
            float it = ipre[gb], ft = fpre[gb];
            float mn = fmaxf(ft + mreg, it);
            s_ig = __expf(it - mn);
            s_fg = __expf(ft + mreg - mn);
            mreg = mn;
        }
        __syncthreads();
        float ig = s_ig, fg = s_fg;
        float vr = vs[rr];
        float qp = 0.f;
#pragma unroll
        for (int j = 0; j < 16; ++j) {
            int c = cbase + j;
            C[j] = fg * C[j] + ig * vr * ks[c];
            qp += C[j] * qs[c];
        }
        partial[tid] = qp;                              // tid == rr*4 + quarter
        if (tid < 64) { nreg = fg * nreg + ig * ks[tid]; nqs[tid] = nreg * qs[tid]; }
        __syncthreads();
        if (tid < 32) {
            float s = nqs[tid] + nqs[tid + 32];
#pragma unroll
            for (int m2 = 16; m2; m2 >>= 1) s += __shfl_xor(s, m2, 32);
            if (tid == 0) s_den = fmaxf(fabsf(s), 1.0f);
        }
        __syncthreads();
        if (tid < 64) {
            float num = partial[tid * 4] + partial[tid * 4 + 1]
                      + partial[tid * 4 + 2] + partial[tid * 4 + 3];
            h[base + tid] = num / s_den;
        }
    }
}

// ---------------------------------------------------------------------------
// ht = o*h; per-head GroupNorm(64); * gn_g + gn_b; * silu(r) -> pre_down
// one wave per (row, head)
// ---------------------------------------------------------------------------
__global__ void gnpre(const float* __restrict__ h, const float* __restrict__ o,
                      const float* __restrict__ r, const float* __restrict__ gg,
                      const float* __restrict__ gb, float* __restrict__ outp, int R)
{
    int wave = blockIdx.x * (blockDim.x >> 5) + (threadIdx.x >> 5);
    if (wave >= R * HEADS) return;
    int lane = threadIdx.x & 31;
    int row = wave >> 2, hd = wave & 3;
    size_t base = (size_t)row * HID + hd * DHEAD;
    float v0 = h[base + lane]      * o[base + lane];
    float v1 = h[base + lane + 32] * o[base + lane + 32];
    float s = v0 + v1, ss = v0 * v0 + v1 * v1;
#pragma unroll
    for (int m = 16; m; m >>= 1) { s += __shfl_xor(s, m, 32); ss += __shfl_xor(ss, m, 32); }
    float mu  = s * (1.f / 64.f);
    float var = ss * (1.f / 64.f) - mu * mu;
    float rs  = rsqrtf(var + 1e-5f);
    int c0 = hd * DHEAD + lane, c1 = c0 + 32;
    float rv0 = r[(size_t)row * HID + c0], rv1 = r[(size_t)row * HID + c1];
    outp[(size_t)row * HID + c0] = ((v0 - mu) * rs * gg[c0] + gb[c0]) * (rv0 * sigmoidf_(rv0));
    outp[(size_t)row * HID + c1] = ((v1 - mu) * rs * gg[c1] + gb[c1]) * (rv1 * sigmoidf_(rv1));
}

// ---------------------------------------------------------------------------
static inline void launch_gemm(const float* A, const float* Amul, const float* W,
                               const float* bias, const float* resid, float* out,
                               int M, int N, int K, float scale, int act, hipStream_t s)
{
    dim3 grid((M + 255) / 256, N / 64);
    if (Amul)
        gemm_wmma<true><<<grid, 256, 0, s>>>(A, Amul, W, bias, resid, out, M, N, K, scale, act);
    else
        gemm_wmma<false><<<grid, 256, 0, s>>>(A, Amul, W, bias, resid, out, M, N, K, scale, act);
}

extern "C" void kernel_launch(void* const* d_in, const int* in_sizes, int n_in,
                              void* d_out, int out_size, void* d_ws, size_t ws_size,
                              hipStream_t stream)
{
    const float* x      = (const float*)d_in[0];
    const int*   maskix = (const int*)  d_in[1];
    const float* W_e    = (const float*)d_in[2];
    const float* b_e    = (const float*)d_in[3];
    const float* ln_g   = (const float*)d_in[4];
    const float* ln_b   = (const float*)d_in[5];
    const float* W1     = (const float*)d_in[6];
    const float* b1     = (const float*)d_in[7];
    const float* W2     = (const float*)d_in[8];
    const float* b2     = (const float*)d_in[9];
    const float* W3     = (const float*)d_in[10];
    const float* b3     = (const float*)d_in[11];
    const float* mln_g  = (const float*)d_in[12];
    const float* mln_b  = (const float*)d_in[13];
    const float* W_ul   = (const float*)d_in[14];
    const float* b_ul   = (const float*)d_in[15];
    const float* W_ur   = (const float*)d_in[16];
    const float* b_ur   = (const float*)d_in[17];
    const float* W_q    = (const float*)d_in[18];
    const float* W_k    = (const float*)d_in[19];
    const float* W_v    = (const float*)d_in[20];
    const float* W_o    = (const float*)d_in[21];
    const float* W_i    = (const float*)d_in[22];
    const float* W_f    = (const float*)d_in[23];
    const float* gn_g   = (const float*)d_in[24];
    const float* gn_b   = (const float*)d_in[25];
    const float* W_down = (const float*)d_in[26];
    const float* b_down = (const float*)d_in[27];

    const int R = RTOT;                          // 8200
    float* ws = (float*)d_ws;
    size_t off = 0;
    float* tok    = ws + off; off += (size_t)R * D_MODEL;
    float* ybuf   = ws + off; off += (size_t)R * D_MODEL;   // y / xn / first_tm
    float* first  = ws + off; off += (size_t)R * D_MODEL;
    float* second = ws + off; off += (size_t)R * D_MODEL;
    float* xu     = ws + off; off += (size_t)R * PD;
    float* rbuf   = ws + off; off += (size_t)R * HID;
    float* qbuf   = ws + off; off += (size_t)R * HID;
    float* kbuf   = ws + off; off += (size_t)R * HID;
    float* vbuf   = ws + off; off += (size_t)R * HID;
    float* obuf   = ws + off; off += (size_t)R * HID;
    float* hbuf   = ws + off; off += (size_t)R * HID;
    float* pre    = ws + off; off += (size_t)R * HID;
    float* ipre   = ws + off; off += (size_t)R * HEADS;
    float* fpre   = ws + off; off += (size_t)R * HEADS;
    (void)ws_size; (void)in_sizes; (void)n_in; (void)out_size;

    // 1. cls rows = 0
    fill_cls<<<(BATCH * D_MODEL + 255) / 256, 256, 0, stream>>>(tok);
    // 2. patch embed GEMM (+ mask epilogue) -> tok rows 1..1024
    {
        dim3 grid(8192 / 256, D_MODEL / 64);
        gemm_embed<<<grid, 256, 0, stream>>>(x, W_e, b_e, maskix, tok);
    }
    // 3. outer LayerNorm
    ln_rows<<<(R + 7) / 8, 256, 0, stream>>>(tok, ln_g, ln_b, ybuf, R);
    // 4/5. dual paths
    launch_gemm(ybuf, nullptr, W1, b1, nullptr, first,  R, D_MODEL, D_MODEL, 1.f, 0, stream);
    launch_gemm(ybuf, nullptr, W2, b2, nullptr, second, R, D_MODEL, D_MODEL, 1.f, 1, stream);
    // 6. mLSTM LN (reuse ybuf as xn)
    ln_rows<<<(R + 7) / 8, 256, 0, stream>>>(first, mln_g, mln_b, ybuf, R);
    // 7/8. up projections
    launch_gemm(ybuf, nullptr, W_ul, b_ul, nullptr, xu,   R, PD,  D_MODEL, 1.f, 0, stream);
    launch_gemm(ybuf, nullptr, W_ur, b_ur, nullptr, rbuf, R, HID, D_MODEL, 1.f, 0, stream);
    // 9-12. q,k,v,o
    launch_gemm(xu, nullptr, W_q, nullptr, nullptr, qbuf, R, HID, PD, 1.f,    0, stream);
    launch_gemm(xu, nullptr, W_k, nullptr, nullptr, kbuf, R, HID, PD, 0.125f, 0, stream);
    launch_gemm(xu, nullptr, W_v, nullptr, nullptr, vbuf, R, HID, PD, 1.f,    0, stream);
    launch_gemm(xu, nullptr, W_o, nullptr, nullptr, obuf, R, HID, PD, 1.f,    2, stream);
    // 13. i/f gates
    ifgate<<<(R * 8 + 255) / 256, 256, 0, stream>>>(xu, W_i, W_f, ipre, fpre, R);
    // 14. recurrent scan
    mlstm_scan<<<BATCH * HEADS, 256, 0, stream>>>(qbuf, kbuf, vbuf, ipre, fpre, hbuf);
    // 15. o*h, GroupNorm, *silu(r)
    gnpre<<<(R * HEADS + 7) / 8, 256, 0, stream>>>(hbuf, obuf, rbuf, gn_g, gn_b, pre, R);
    // 16. down projection + residual(first) -> first_tm (reuse ybuf)
    launch_gemm(pre, nullptr, W_down, b_down, first, ybuf, R, D_MODEL, HID, 1.f, 0, stream);
    // 17. (first_tm * second) @ W3 + b3 + tok -> out
    launch_gemm(ybuf, second, W3, b3, tok, (float*)d_out, R, D_MODEL, D_MODEL, 1.f, 0, stream);
}
